// InvariantPointAttention_27118423507357
// MI455X (gfx1250) — matrix operand order repaired and checked by reading, not compiled, not run on hardware
//
#include <hip/hip_runtime.h>
#include <math.h>

// ---------------------------------------------------------------------------
// Invariant Point Attention forward for MI455X (gfx1250), wave32 + WMMA +
// async global->LDS streaming for the HBM-bound z tensor.
// B=1, N=768, CS=384, CZ=128, CH=16, H=12, PQ=4, PV=8.
// ---------------------------------------------------------------------------

#define N_RES 768
#define CS_   384
#define CZ_   128
#define CH_   16
#define H_    12
#define PQ_   4
#define PV_   8
#define CAT_  2112   // H*(CZ+CH+PV*4) = 12*176

typedef __attribute__((ext_vector_type(16))) __bf16 bf16x16;
typedef __attribute__((ext_vector_type(8)))  float  f32x8;

union FragBF { bf16x16 v; unsigned int u[8]; };

#if __has_builtin(__builtin_amdgcn_global_load_async_to_lds_b128) && \
    __has_builtin(__builtin_amdgcn_s_wait_asynccnt)
#define HAS_ASYNC_LDS 1
#else
#define HAS_ASYNC_LDS 0
#endif

#if HAS_ASYNC_LDS
// 16-byte async DMA global -> LDS (ASYNCcnt-tracked).
// The builtin takes int4* pointers: addrspace(1) source, addrspace(3) dest.
// Integer detours for the address-space casts: the generic LDS address
// truncated to 32 bits == LDS offset (aperture rule addr[31:0]).
typedef int v4i_ __attribute__((vector_size(16)));
typedef __attribute__((address_space(1))) v4i_ gv4i_t;
typedef __attribute__((address_space(3))) v4i_ lv4i_t;

__device__ __forceinline__ void async_copy16(const void* g, void* l) {
    __builtin_amdgcn_global_load_async_to_lds_b128(
        (gv4i_t*)(unsigned long long)g,
        (lv4i_t*)(unsigned int)(unsigned long long)l,
        0, 0);
}
#endif

__device__ __forceinline__ unsigned int f2bf(float f) {
    unsigned int x = __float_as_uint(f);
    unsigned int r = x + 0x7FFFu + ((x >> 16) & 1u);   // round-to-nearest-even
    return r >> 16;
}

__device__ __forceinline__ float wave_max(float v) {
    for (int o = 16; o > 0; o >>= 1) v = fmaxf(v, __shfl_xor(v, o));
    return v;
}
__device__ __forceinline__ float wave_sum(float v) {
    for (int o = 16; o > 0; o >>= 1) v += __shfl_xor(v, o);
    return v;
}

// A fragment (16x32, ISA 7.12.2 layout) gathered from an f32 LDS tile,
// packed to bf16 pairs in registers.
__device__ __forceinline__ FragBF gather_a_f32(const float* As, int lane) {
    FragBF a;
    const int half = lane >> 4;
    const int m    = lane & 15;
#pragma unroll
    for (int v = 0; v < 8; ++v) {
        int k = (v < 4) ? (half * 8 + 2 * v) : (16 + half * 8 + 2 * (v - 4));
        a.u[v] = f2bf(As[m * 32 + k]) | (f2bf(As[m * 32 + k + 1]) << 16);
    }
    return a;
}

// B fragment (32x16, column ncol per lane) from an f32 LDS tile [32][128].
__device__ __forceinline__ FragBF gather_b_f32(const float* Bs, int lane, int ncol) {
    FragBF b;
#pragma unroll
    for (int v = 0; v < 8; ++v) {
        int k = (v < 4) ? ((lane >> 4) * 8 + 2 * v) : (16 + (lane >> 4) * 8 + 2 * (v - 4));
        b.u[v] = f2bf(Bs[k * 128 + ncol]) | (f2bf(Bs[(k + 1) * 128 + ncol]) << 16);
    }
    return b;
}

// ---------------------------------------------------------------------------
// Generic GEMM:  C[M x Nc] = A[M x K] (lda) @ Bm[K x Nc] (ldb) + bias
// Preconditions (guaranteed by all launches): M % 16 == 0, K % 32 == 0.
// block = 256 threads (8 waves). Block tile: 16 rows x 128 cols.
// grid.x = M/16, grid.y = ceil(Nc/128). A tile streamed by async DMA.
// ---------------------------------------------------------------------------
__global__ void gemm_bf16_wmma(const float* __restrict__ A, int lda,
                               const float* __restrict__ Bm, int ldb,
                               const float* __restrict__ bias,
                               float* __restrict__ C, int ldc,
                               int M, int K, int Nc) {
    __shared__ float Asf[16 * 32];
    __shared__ float Bsf[32 * 128];

    const int tid  = threadIdx.x;
    const int lane = tid & 31;
    const int wave = tid >> 5;
    const int row0 = blockIdx.x * 16;
    const int n0   = blockIdx.y * 128;

    f32x8 acc = {0.f, 0.f, 0.f, 0.f, 0.f, 0.f, 0.f, 0.f};

    for (int k0 = 0; k0 < K; k0 += 32) {
#if HAS_ASYNC_LDS
        if (tid < 128) {   // 128 x 16B = full 16x32 f32 A tile, no bounds needed
            int r = tid >> 3, c4 = (tid & 7) * 4;
            async_copy16(A + (long)(row0 + r) * lda + k0 + c4, &Asf[r * 32 + c4]);
        }
#else
        for (int t = tid; t < 512; t += 256) {
            int r = t >> 5, kk = t & 31;
            Asf[t] = A[(long)(row0 + r) * lda + k0 + kk];
        }
#endif
        for (int t = tid; t < 32 * 128; t += 256) {
            int kk = t >> 7, n = t & 127;
            int gn = n0 + n;
            Bsf[t] = (gn < Nc) ? Bm[(long)(k0 + kk) * ldb + gn] : 0.f;
        }
#if HAS_ASYNC_LDS
        __builtin_amdgcn_s_wait_asynccnt(0);
#endif
        __syncthreads();

        FragBF a = gather_a_f32(Asf, lane);
        FragBF b = gather_b_f32(Bsf, lane, wave * 16 + (lane & 15));
        acc = __builtin_amdgcn_wmma_f32_16x16x32_bf16(
            false, a.v, false, b.v, (short)0, acc, false, false);
        __syncthreads();
    }

    // C/D layout: row = v + 8*(lane>=16), col = lane&15
    const int half = lane >> 4;
    const int gn   = n0 + wave * 16 + (lane & 15);
    if (gn < Nc) {
        float bv = bias ? bias[gn] : 0.f;
#pragma unroll
        for (int v = 0; v < 8; ++v) {
            int gr = row0 + v + 8 * half;
            C[(long)gr * ldc + gn] = acc[v] + bv;
        }
    }
}

// ---------------------------------------------------------------------------
// Frame application: pts[d] = loc[i, d*third + m]; out = rot[i] @ pts + trans[i]
// ---------------------------------------------------------------------------
__global__ void frames_q(const float* __restrict__ loc, const float* __restrict__ rot,
                         const float* __restrict__ trans, float* __restrict__ q_pts) {
    int i = blockIdx.x;
    const float* R = rot + i * 9;
    const float* T = trans + i * 3;
    for (int m = threadIdx.x; m < H_ * PQ_; m += blockDim.x) {
        float p0 = loc[i * 144 + 0 * 48 + m];
        float p1 = loc[i * 144 + 1 * 48 + m];
        float p2 = loc[i * 144 + 2 * 48 + m];
        float* o = q_pts + ((long)i * H_ * PQ_ + m) * 3;
        o[0] = R[0] * p0 + R[1] * p1 + R[2] * p2 + T[0];
        o[1] = R[3] * p0 + R[4] * p1 + R[5] * p2 + T[1];
        o[2] = R[6] * p0 + R[7] * p1 + R[8] * p2 + T[2];
    }
}

__global__ void frames_kv(const float* __restrict__ loc, const float* __restrict__ rot,
                          const float* __restrict__ trans,
                          float* __restrict__ k_pts, float* __restrict__ v_pts) {
    int i = blockIdx.x;
    const float* R = rot + i * 9;
    const float* T = trans + i * 3;
    for (int m = threadIdx.x; m < H_ * (PQ_ + PV_); m += blockDim.x) {
        float p0 = loc[i * 432 + 0 * 144 + m];
        float p1 = loc[i * 432 + 1 * 144 + m];
        float p2 = loc[i * 432 + 2 * 144 + m];
        float x = R[0] * p0 + R[1] * p1 + R[2] * p2 + T[0];
        float y = R[3] * p0 + R[4] * p1 + R[5] * p2 + T[1];
        float z = R[6] * p0 + R[7] * p1 + R[8] * p2 + T[2];
        int h = m / 12, pp = m % 12;
        float* o;
        if (pp < PQ_) o = k_pts + (((long)i * H_ + h) * PQ_ + pp) * 3;
        else          o = v_pts + (((long)i * H_ + h) * PV_ + (pp - PQ_)) * 3;
        o[0] = x; o[1] = y; o[2] = z;
    }
}

// ---------------------------------------------------------------------------
// Fused logits (qk + pair bias + point attention) + softmax.
// One wave per (h, i) row; 8 waves / block. grid.x = H*N/8.
// ---------------------------------------------------------------------------
__global__ void logits_softmax(const float* __restrict__ q_buf,
                               const float* __restrict__ kv_buf,
                               const float* __restrict__ q_pts,
                               const float* __restrict__ k_pts,
                               const float* __restrict__ b_buf,
                               const float* __restrict__ head_w,
                               float* __restrict__ a_buf) {
    const int lane = threadIdx.x & 31;
    const int wave = threadIdx.x >> 5;
    const int idx  = blockIdx.x * 8 + wave;           // (h, i) row
    const int h = idx / N_RES;
    const int i = idx % N_RES;

    const float scale_qk = rsqrtf(3.0f * CH_);        // 1/sqrt(48)
    const float scale_b  = sqrtf(1.0f / 3.0f);
    const float hw = logf(1.0f + __expf(head_w[h])) * sqrtf(1.0f / 54.0f);

    float qc[CH_];
#pragma unroll
    for (int c = 0; c < CH_; ++c) qc[c] = q_buf[(long)i * (H_ * CH_) + h * CH_ + c];
    float qp[PQ_][3];
#pragma unroll
    for (int p = 0; p < PQ_; ++p)
#pragma unroll
        for (int d = 0; d < 3; ++d)
            qp[p][d] = q_pts[(((long)i * H_ + h) * PQ_ + p) * 3 + d];

    float lg[N_RES / 32];
#pragma unroll 4
    for (int jj = 0; jj < N_RES / 32; ++jj) {
        int j = jj * 32 + lane;
        const float* kr = kv_buf + (long)j * (2 * H_ * CH_) + h * (2 * CH_);
        float dot = 0.f;
#pragma unroll
        for (int c = 0; c < CH_; ++c) dot += qc[c] * kr[c];
        float d2 = 0.f;
        const float* kp = k_pts + ((long)j * H_ + h) * PQ_ * 3;
#pragma unroll
        for (int p = 0; p < PQ_; ++p)
#pragma unroll
            for (int d = 0; d < 3; ++d) {
                float df = qp[p][d] - kp[p * 3 + d];
                d2 += df * df;
            }
        float bij = b_buf[((long)i * N_RES + j) * H_ + h];
        lg[jj] = dot * scale_qk + scale_b * bij - 0.5f * hw * d2;
    }

    float mx = -3.0e38f;
#pragma unroll
    for (int jj = 0; jj < N_RES / 32; ++jj) mx = fmaxf(mx, lg[jj]);
    mx = wave_max(mx);

    float sum = 0.f;
#pragma unroll
    for (int jj = 0; jj < N_RES / 32; ++jj) {
        lg[jj] = __expf(lg[jj] - mx);
        sum += lg[jj];
    }
    sum = wave_sum(sum);
    float inv = 1.0f / sum;

    float* arow = a_buf + (long)h * N_RES * N_RES + (long)i * N_RES;
#pragma unroll 4
    for (int jj = 0; jj < N_RES / 32; ++jj)
        arow[jj * 32 + lane] = lg[jj] * inv;
}

// ---------------------------------------------------------------------------
// o_pair[i,h,c] = sum_j a[h,i,j] * z[i,j,c]. Batched per-i WMMA:
// per block i: (16pad x 768) @ (768 x 128). 8 waves, one 16-col tile each.
// z tile (16 KB) double-buffered via async global->LDS DMA.
// ---------------------------------------------------------------------------
__global__ void opair_wmma(const float* __restrict__ a_buf,
                           const float* __restrict__ z,
                           float* __restrict__ o_pair) {
    __shared__ float Asf[16 * 32];
    __shared__ float Bsf[2][32 * 128];

    const int i    = blockIdx.x;
    const int tid  = threadIdx.x;
    const int lane = tid & 31;
    const int wave = tid >> 5;

    f32x8 acc = {0.f, 0.f, 0.f, 0.f, 0.f, 0.f, 0.f, 0.f};

#if HAS_ASYNC_LDS
    {   // prologue: stream tile 0 (4 async instructions per wave)
        const float* zt = z + (long)i * N_RES * CZ_;
#pragma unroll
        for (int s = 0; s < 4; ++s) {
            int t = tid + s * 256;
            int kk = t >> 5, c = (t & 31) * 4;
            async_copy16(zt + kk * CZ_ + c, &Bsf[0][kk * 128 + c]);
        }
    }
#endif

    for (int k0 = 0; k0 < N_RES; k0 += 32) {
        const int buf = (k0 >> 5) & 1;
#if HAS_ASYNC_LDS
        if (k0 + 32 < N_RES) {
            const float* zt = z + ((long)i * N_RES + k0 + 32) * CZ_;
#pragma unroll
            for (int s = 0; s < 4; ++s) {
                int t = tid + s * 256;
                int kk = t >> 5, c = (t & 31) * 4;
                async_copy16(zt + kk * CZ_ + c, &Bsf[buf ^ 1][kk * 128 + c]);
            }
            __builtin_amdgcn_s_wait_asynccnt(4);   // current tile's 4 done
        } else {
            __builtin_amdgcn_s_wait_asynccnt(0);
        }
#else
        for (int t = tid; t < 32 * 128; t += 256) {
            int kk = t >> 7, n = t & 127;
            Bsf[buf][t] = z[((long)i * N_RES + k0 + kk) * CZ_ + n];
        }
#endif
        // A tile: 12 live rows of the attention matrix (sync, small)
        for (int t = tid; t < 512; t += 256) {
            int m = t >> 5, kk = t & 31;
            Asf[t] = (m < H_)
                ? a_buf[(long)m * N_RES * N_RES + (long)i * N_RES + k0 + kk]
                : 0.f;
        }
        __syncthreads();

        FragBF a = gather_a_f32(Asf, lane);
        FragBF b = gather_b_f32(Bsf[buf], lane, wave * 16 + (lane & 15));
        acc = __builtin_amdgcn_wmma_f32_16x16x32_bf16(
            false, a.v, false, b.v, (short)0, acc, false, false);
        __syncthreads();
    }

    const int half = lane >> 4;
    const int nn   = wave * 16 + (lane & 15);
#pragma unroll
    for (int v = 0; v < 8; ++v) {
        int row = v + 8 * half;                       // head index
        if (row < H_)
            o_pair[(long)i * (H_ * CZ_) + row * CZ_ + nn] = acc[v];
    }
}

// ---------------------------------------------------------------------------
// Inverse frame, point norms, concat.
// cat = [o(192) | x(96) | y(96) | z(96) | norm(96) | o_pair(1536)]
// ---------------------------------------------------------------------------
__global__ void finalize_cat(const float* __restrict__ o_buf,
                             const float* __restrict__ o_pt,
                             const float* __restrict__ o_pair,
                             const float* __restrict__ rot,
                             const float* __restrict__ trans,
                             float* __restrict__ cat) {
    int i = blockIdx.x;
    const float* R = rot + i * 9;
    const float* T = trans + i * 3;
    for (int col = threadIdx.x; col < CAT_; col += blockDim.x) {
        float val;
        if (col < 192) {
            val = o_buf[(long)i * 192 + col];
        } else if (col < 576) {
            int g = (col - 192) / 96;                 // 0=x 1=y 2=z 3=norm
            int m = (col - 192) % 96;                 // h*PV + p
            int h = m >> 3, p = m & 7;
            const float* gp = o_pt + (long)i * 288 + h * 24 + p * 3;
            float v0 = gp[0] - T[0], v1 = gp[1] - T[1], v2 = gp[2] - T[2];
            float lx = R[0] * v0 + R[3] * v1 + R[6] * v2;   // R^T * v
            float ly = R[1] * v0 + R[4] * v1 + R[7] * v2;
            float lz = R[2] * v0 + R[5] * v1 + R[8] * v2;
            val = (g == 0) ? lx : (g == 1) ? ly : (g == 2) ? lz
                : sqrtf(lx * lx + ly * ly + lz * lz + 1e-8f);
        } else {
            val = o_pair[(long)i * (H_ * CZ_) + (col - 576)];
        }
        cat[(long)i * CAT_ + col] = val;
    }
}

// ---------------------------------------------------------------------------
extern "C" void kernel_launch(void* const* d_in, const int* in_sizes, int n_in,
                              void* d_out, int out_size, void* d_ws, size_t ws_size,
                              hipStream_t stream) {
    const float* s     = (const float*)d_in[0];
    const float* z     = (const float*)d_in[1];
    const float* rot   = (const float*)d_in[2];
    const float* trans = (const float*)d_in[3];
    const float* Wq    = (const float*)d_in[4];
    const float* bq    = (const float*)d_in[5];
    const float* Wkv   = (const float*)d_in[6];
    const float* bkv   = (const float*)d_in[7];
    const float* Wqp   = (const float*)d_in[8];
    const float* bqp   = (const float*)d_in[9];
    const float* Wkvp  = (const float*)d_in[10];
    const float* bkvp  = (const float*)d_in[11];
    const float* Wb    = (const float*)d_in[12];
    const float* bb    = (const float*)d_in[13];
    const float* hwp   = (const float*)d_in[14];
    const float* Wout  = (const float*)d_in[15];
    const float* bout  = (const float*)d_in[16];
    float* out = (float*)d_out;

    float* w = (float*)d_ws;
    size_t off = 0;
    auto alloc = [&](size_t n) { float* p = w + off; off += n; return p; };

    float* q_buf   = alloc((size_t)N_RES * H_ * CH_);            // 147456
    float* kv_buf  = alloc((size_t)N_RES * 2 * H_ * CH_);        // 294912
    float* qp_loc  = alloc((size_t)N_RES * 144);
    float* kvp_loc = alloc((size_t)N_RES * 432);
    float* q_pts   = alloc((size_t)N_RES * H_ * PQ_ * 3);
    float* k_pts   = alloc((size_t)N_RES * H_ * PQ_ * 3);
    float* v_pts   = alloc((size_t)N_RES * H_ * PV_ * 3);
    float* b_buf   = alloc((size_t)N_RES * N_RES * H_);          // 7.08M
    float* a_buf   = alloc((size_t)H_ * N_RES * N_RES);          // 7.08M
    float* o_buf   = alloc((size_t)N_RES * H_ * CH_);
    float* o_pt    = alloc((size_t)N_RES * H_ * PV_ * 3);
    float* o_pair  = alloc((size_t)N_RES * H_ * CZ_);
    float* cat     = alloc((size_t)N_RES * CAT_);

    dim3 blk(256);
    auto gemm = [&](const float* A, int lda, const float* Bm, int ldb,
                    const float* bias, float* C, int ldc, int M, int K, int Nc) {
        dim3 grid((M + 15) / 16, (Nc + 127) / 128);
        gemm_bf16_wmma<<<grid, blk, 0, stream>>>(A, lda, Bm, ldb, bias, C, ldc, M, K, Nc);
    };

    // --- projections: s @ {Wq, Wkv, Wqp, Wkvp} ---
    gemm(s, CS_, Wq,   H_ * CH_,      bq,   q_buf,   H_ * CH_,      N_RES, CS_, H_ * CH_);
    gemm(s, CS_, Wkv,  2 * H_ * CH_,  bkv,  kv_buf,  2 * H_ * CH_,  N_RES, CS_, 2 * H_ * CH_);
    gemm(s, CS_, Wqp,  144,           bqp,  qp_loc,  144,           N_RES, CS_, 144);
    gemm(s, CS_, Wkvp, 432,           bkvp, kvp_loc, 432,           N_RES, CS_, 432);

    // --- frames ---
    frames_q<<<dim3(N_RES), dim3(64), 0, stream>>>(qp_loc, rot, trans, q_pts);
    frames_kv<<<dim3(N_RES), dim3(160), 0, stream>>>(kvp_loc, rot, trans, k_pts, v_pts);

    // --- pair bias: b = z @ Wb + bb (589824 x 128 x 12), z streamed as A ---
    gemm(z, CZ_, Wb, H_, bb, b_buf, H_, N_RES * N_RES, CZ_, H_);

    // --- logits + softmax ---
    logits_softmax<<<dim3(H_ * N_RES / 8), blk, 0, stream>>>(
        q_buf, kv_buf, q_pts, k_pts, b_buf, hwp, a_buf);

    // --- per-head o = a @ v and o_pt = a @ v_pts ---
    for (int h = 0; h < H_; ++h) {
        gemm(a_buf + (size_t)h * N_RES * N_RES, N_RES,
             kv_buf + h * 2 * CH_ + CH_, 2 * H_ * CH_, nullptr,
             o_buf + h * CH_, H_ * CH_, N_RES, N_RES, CH_);
        gemm(a_buf + (size_t)h * N_RES * N_RES, N_RES,
             v_pts + h * PV_ * 3, H_ * PV_ * 3, nullptr,
             o_pt + h * PV_ * 3, H_ * PV_ * 3, N_RES, N_RES, PV_ * 3);
    }

    // --- o_pair (batched per residue, async double-buffered z stream) ---
    opair_wmma<<<dim3(N_RES), blk, 0, stream>>>(a_buf, z, o_pair);

    // --- inverse frames + norms + concat ---
    finalize_cat<<<dim3(N_RES), blk, 0, stream>>>(o_buf, o_pt, o_pair, rot, trans, cat);

    // --- output projection ---
    gemm(cat, CAT_, Wout, CS_, bout, out, CS_, N_RES, CAT_, CS_);
}
